// TreeEnergyLossBinarySAM_88802743812179
// MI455X (gfx1250) — matrix operand order repaired and checked by reading, not compiled, not run on hardware
//
#include <hip/hip_runtime.h>
#include <cstdint>

#define DEV __device__ __forceinline__

constexpr int BATCH = 4;
constexpr int H = 80, W = 80;
constexpr int NPIX = H * W;                // 6400
constexpr int NEH  = H * (W - 1);          // 6320 horizontal edges
constexpr int M_EDGES = NEH + W * (H - 1); // 12640
constexpr int CLOW = 3;
constexpr int CHI  = 768;
constexpr float SIGMA_F = 0.002f;
constexpr float LOSS_WEIGHT = 0.4f;
constexpr int N_JUMP   = 14;
constexpr int N_ROUNDS = 16;
constexpr int TILES_E = M_EDGES / 16;      // 790 (exact)
constexpr int TILES_N = NPIX / 16;         // 400 (exact)

typedef _Float16 v16h __attribute__((ext_vector_type(16)));
typedef float    v8f  __attribute__((ext_vector_type(8)));
union V16U { v16h v; _Float16 f[16]; };
union V8FU { v8f  v; float    f[8]; };

DEV void edge_uv(int e, int& u, int& v) {
  if (e < NEH) { int r = e / (W - 1); int c = e - r * (W - 1); u = r * W + c; v = u + 1; }
  else         { int e2 = e - NEH; u = e2; v = e2 + W; }
}

// extract acc.f[lane&7] without dynamic vector indexing (no scratch spill)
DEV float diag_select(const V8FU& acc, int lane) {
  int r = lane & 7;
  float d = acc.f[0];
#pragma unroll
  for (int j = 1; j < 8; ++j) d = (r == j) ? acc.f[j] : d;
  return d;
}

// ---------------- resize low_feats 320x320 -> 80x80 (bilinear, scale 1/4:
// sample coord 4o+1.5 => mean of the 2x2 block at rows/cols 4o+1..4o+2) -----
__global__ void resize_low_kernel(const float* __restrict__ low, float* __restrict__ lowr) {
  int idx = blockIdx.x * blockDim.x + threadIdx.x;
  if (idx >= BATCH * CLOW * NPIX) return;
  int pix = idx % NPIX, bc = idx / NPIX;
  int r = pix / W, c = pix % W;
  const float* src = low + (size_t)bc * (4 * H) * (4 * W);
  int r0 = 4 * r + 1, c0 = 4 * c + 1, ld = 4 * W;
  float s = src[r0 * ld + c0] + src[r0 * ld + c0 + 1] +
            src[(r0 + 1) * ld + c0] + src[(r0 + 1) * ld + c0 + 1];
  lowr[idx] = 0.25f * s;
}

// ---------------- 3-channel edge weights (tree 1) --------------------------
__global__ void w_low_kernel(const float* __restrict__ lowr, float* __restrict__ w1) {
  int idx = blockIdx.x * blockDim.x + threadIdx.x;
  if (idx >= BATCH * M_EDGES) return;
  int b = idx / M_EDGES, e = idx % M_EDGES;
  const float* LF = lowr + (size_t)b * CLOW * NPIX;
  int u, v; edge_uv(e, u, v);
  float s = 0.f;
#pragma unroll
  for (int c = 0; c < CLOW; ++c) {
    float d = LF[c * NPIX + u] - LF[c * NPIX + v];
    s += d * d;
  }
  w1[idx] = s;
}

// ---------------- 3-channel node->parent distances (tree 1) ----------------
__global__ void dist_low_kernel(const float* __restrict__ lowr, const int* __restrict__ par_all,
                                float* __restrict__ dist_all) {
  int idx = blockIdx.x * blockDim.x + threadIdx.x;
  if (idx >= BATCH * NPIX) return;
  int b = idx / NPIX, i = idx % NPIX;
  const float* LF = lowr + (size_t)b * CLOW * NPIX;
  int p = par_all[idx];
  float s = 0.f;
#pragma unroll
  for (int c = 0; c < CLOW; ++c) {
    float d = LF[c * NPIX + i] - LF[c * NPIX + p];
    s += d * d;
  }
  dist_all[idx] = s;
}

// ---------------- WMMA core: diag(Delta * Delta^T), K accumulated in f32 ---
// One wave32 per 16-row tile. Per 32-channel chunk the wave stages the
// 16x32 f16 difference tile in LDS (coalesced: half-wave shares a channel
// across 16 near-consecutive nodes), then builds operands from LDS:
//   A (16x32, lane->M=lane&15): contiguous K runs [base..base+7] and
//     [16+base..16+base+7], base = (lane>=16)*8  -> two ds b128 loads
//   B (32x16, lane->N=lane&15): same K runs as A [base..base+7] and
//     [16+base..16+base+7], base = (lane>=16)*8
// Feeding Delta rows through the B mapping yields B = Delta^T, so
// D = Delta * Delta^T and diag(D) are the 16 squared norms.
DEV void gram_diag_tile(const float* __restrict__ rowu, const float* __restrict__ rowv,
                        _Float16* __restrict__ tile, int lane, V8FU& acc) {
  const int m = lane & 15;
  const int khalf = lane >> 4;  // this lane stages channels k = 2t + khalf
#pragma unroll
  for (int i = 0; i < 8; ++i) acc.f[i] = 0.f;
  for (int k0 = 0; k0 < CHI; k0 += 32) {
    if (k0 + 32 < CHI)  // prefetch next chunk (global_prefetch_b8)
      __builtin_prefetch(rowu + (size_t)(k0 + 32 + khalf) * NPIX, 0, 1);
#pragma unroll
    for (int t = 0; t < 16; ++t) {
      int k = 2 * t + khalf;
      size_t o = (size_t)(k0 + k) * NPIX;
      tile[m * 32 + k] = (_Float16)(rowu[o] - rowv[o]);
    }
    __syncthreads();
    V16U A, Bm;
    const int abase = (lane & 16) ? 8 : 0;
    const int bbase = (lane & 16) ? 16 : 0;
#pragma unroll
    for (int e = 0; e < 8; ++e) {
      A.f[e]     = tile[m * 32 + abase + e];
      A.f[8 + e] = tile[m * 32 + 16 + abase + e];
    }
#pragma unroll
    for (int e = 0; e < 16; ++e) Bm.f[e] = tile[m * 32 + (e < 8 ? bbase / 2 + e : 16 + bbase / 2 + (e - 8))];
    acc.v = __builtin_amdgcn_wmma_f32_16x16x32_f16(false, A.v, false, Bm.v,
                                                   (short)0, acc.v, false, false);
    __syncthreads();
  }
}

// 768-channel edge weights (tree 2): one wave per 16-edge tile.
__global__ void w_high_wmma_kernel(const float* __restrict__ high, float* __restrict__ w2) {
  __shared__ __align__(16) _Float16 tile[16 * 32];
  const int lane = threadIdx.x;            // wave32
  const int b = blockIdx.y;
  const int e0 = blockIdx.x * 16;
  const float* HF = high + (size_t)b * CHI * NPIX;
  int u, v; edge_uv(e0 + (lane & 15), u, v);
  V8FU acc;
  gram_diag_tile(HF + u, HF + v, tile, lane, acc);
  float dval = diag_select(acc, lane);
  float* W2 = w2 + (size_t)b * M_EDGES;
  if (lane < 8)        W2[e0 + lane]        = dval;   // M=N=0..7
  else if (lane >= 24) W2[e0 + (lane - 16)] = dval;   // M=N=8..15
}

// 768-channel node->parent distances (tree 2): one wave per 16-node tile.
__global__ void dist_high_wmma_kernel(const float* __restrict__ high,
                                      const int* __restrict__ par_all,
                                      float* __restrict__ dist_all) {
  __shared__ __align__(16) _Float16 tile[16 * 32];
  const int lane = threadIdx.x;            // wave32
  const int b = blockIdx.y;
  const int i0 = blockIdx.x * 16;
  const float* HF = high + (size_t)b * CHI * NPIX;
  const int* par = par_all + (size_t)b * NPIX;
  int row = i0 + (lane & 15);
  int prow = par[row];
  V8FU acc;
  gram_diag_tile(HF + row, HF + prow, tile, lane, acc);
  float dval = diag_select(acc, lane);
  float* D = dist_all + (size_t)b * NPIX;
  if (lane < 8)        D[i0 + lane]        = dval;
  else if (lane >= 24) D[i0 + (lane - 16)] = dval;
}

// ---------------- Boruvka MST: one 1024-thread block per sample, all graph
// state resident in 128 KB of LDS (CDNA5: 320 KB/WGP) ----------------------
__global__ void boruvka_kernel(const float* __restrict__ w_all, int* __restrict__ inmst_all) {
  extern __shared__ char smemraw[];
  int*      comp = (int*)smemraw;            // n
  int*      parA = comp + NPIX;              // n
  int*      parB = parA + NPIX;              // n
  unsigned* minw = (unsigned*)(parB + NPIX); // n (float bits; nonneg => uint-ordered)
  int*      sel  = (int*)(minw + NPIX);      // n
  const int b = blockIdx.x;
  const float* wgt = w_all + (size_t)b * M_EDGES;
  int* inmst = inmst_all + (size_t)b * M_EDGES;
  const int tid = threadIdx.x, nt = blockDim.x;

  for (int i = tid; i < NPIX; i += nt) comp[i] = i;
  for (int e = tid; e < M_EDGES; e += nt) inmst[e] = 0;
  __syncthreads();

  for (int round = 0; round < N_ROUNDS; ++round) {
    int* par = parA; int* tmp = parB;
    for (int i = tid; i < NPIX; i += nt) { minw[i] = 0x7F800000u; sel[i] = M_EDGES; }
    __syncthreads();
    for (int e = tid; e < M_EDGES; e += nt) {
      int u, v; edge_uv(e, u, v);
      int cu = comp[u], cv = comp[v];
      if (cu != cv) {
        unsigned wb = __float_as_uint(wgt[e]);
        atomicMin(&minw[cu], wb);
        atomicMin(&minw[cv], wb);
      }
    }
    __syncthreads();
    for (int e = tid; e < M_EDGES; e += nt) {
      int u, v; edge_uv(e, u, v);
      int cu = comp[u], cv = comp[v];
      if (cu != cv) {
        float wv = wgt[e];
        if (wv <= __uint_as_float(minw[cu])) atomicMin(&sel[cu], e);
        if (wv <= __uint_as_float(minw[cv])) atomicMin(&sel[cv], e);
      }
    }
    __syncthreads();
    for (int i = tid; i < NPIX; i += nt) {
      int s = sel[i];
      if (s < M_EDGES) {
        inmst[s] = 1;
        int u, v; edge_uv(s, u, v);
        int ecu = comp[u], ecv = comp[v];
        par[i] = (i == ecu) ? ecv : ecu;
      } else par[i] = i;
    }
    __syncthreads();
    // break mutual-selection 2-cycles
    for (int i = tid; i < NPIX; i += nt) {
      int p = par[i];
      tmp[i] = (par[p] == i && i < p) ? i : p;
    }
    __syncthreads();
    { int* t = par; par = tmp; tmp = t; }
    // pointer jumping
    for (int j = 0; j < N_JUMP; ++j) {
      for (int i = tid; i < NPIX; i += nt) tmp[i] = par[par[i]];
      __syncthreads();
      int* t = par; par = tmp; tmp = t;
    }
    for (int i = tid; i < NPIX; i += nt) comp[i] = par[comp[i]];
    __syncthreads();
  }
}

// ---------------- BFS rooting at node 0: persistent block per sample -------
__global__ void bfs_kernel(const int* __restrict__ inmst_all, int* __restrict__ par_all,
                           int* __restrict__ dep_all, int* __restrict__ maxd_all) {
  extern __shared__ char smemraw[];
  int* visc  = (int*)smemraw;   // n
  int* visn  = visc + NPIX;     // n
  int* flags = visn + NPIX;     // [0]=changed, [1]=maxdepth
  const int b = blockIdx.x;
  const int* inmst = inmst_all + (size_t)b * M_EDGES;
  int* par = par_all + (size_t)b * NPIX;
  int* dep = dep_all + (size_t)b * NPIX;
  const int tid = threadIdx.x, nt = blockDim.x;
  for (int i = tid; i < NPIX; i += nt) {
    int v0 = (i == 0) ? 1 : 0;
    visc[i] = v0; visn[i] = v0; par[i] = 0; dep[i] = 0;
  }
  if (tid == 0) { flags[0] = 1; flags[1] = 0; }
  __syncthreads();
  for (int it = 0; it < 2 * NPIX && flags[0]; ++it) {
    __syncthreads();
    if (tid == 0) flags[0] = 0;
    __syncthreads();
    for (int e = tid; e < M_EDGES; e += nt) {
      if (!inmst[e]) continue;
      int u, v; edge_uv(e, u, v);
      int vu = visc[u], vv = visc[v];
      if (vu && !vv) { par[v] = u; dep[v] = dep[u] + 1; visn[v] = 1; flags[0] = 1; }
      else if (vv && !vu) { par[u] = v; dep[u] = dep[v] + 1; visn[u] = 1; flags[0] = 1; }
    }
    __syncthreads();
    for (int i = tid; i < NPIX; i += nt) visc[i] = visn[i];
    __syncthreads();
  }
  for (int i = tid; i < NPIX; i += nt) atomicMax(&flags[1], dep[i]);
  __syncthreads();
  if (tid == 0) maxd_all[b] = flags[1];
}

// ---------------- normalized tree filter: leaf->root then root->leaf DP ----
__global__ void treefilter_kernel(const float* __restrict__ feat_all,
                                  const float* __restrict__ dist_all,
                                  const int* __restrict__ par_all,
                                  const int* __restrict__ dep_all,
                                  const int* __restrict__ maxd_all,
                                  float* __restrict__ out_all) {
  extern __shared__ char smemraw[];
  float* A0 = (float*)smemraw;  // n
  float* A1 = A0 + NPIX;        // n
  float* S0 = A1 + NPIX;        // n
  float* S1 = S0 + NPIX;        // n
  float* ew = S1 + NPIX;        // n
  const int b = blockIdx.x;
  const float* feat = feat_all + (size_t)b * NPIX;
  const float* dist = dist_all + (size_t)b * NPIX;
  const int* par = par_all + (size_t)b * NPIX;
  const int* dep = dep_all + (size_t)b * NPIX;
  float* outp = out_all + (size_t)b * NPIX;
  const int tid = threadIdx.x, nt = blockDim.x;
  const int maxd = maxd_all[b];
  for (int i = tid; i < NPIX; i += nt) {
    ew[i] = (dep[i] > 0) ? __expf(-dist[i] / SIGMA_F) : 0.0f;
    A0[i] = feat[i];
    A1[i] = 1.0f;
  }
  __syncthreads();
  for (int d = maxd; d >= 1; --d) {          // subtree aggregation
    for (int i = tid; i < NPIX; i += nt) {
      if (dep[i] == d) {
        int p = par[i]; float e = ew[i];
        atomicAdd(&A0[p], e * A0[i]);
        atomicAdd(&A1[p], e * A1[i]);
      }
    }
    __syncthreads();
  }
  for (int i = tid; i < NPIX; i += nt) { S0[i] = A0[i]; S1[i] = A1[i]; }
  __syncthreads();
  for (int d = 1; d <= maxd; ++d) {          // through-parent completion
    for (int i = tid; i < NPIX; i += nt) {
      if (dep[i] == d) {
        int p = par[i]; float e = ew[i];
        S0[i] = A0[i] + e * (S0[p] - e * A0[i]);
        S1[i] = A1[i] + e * (S1[p] - e * A1[i]);
      }
    }
    __syncthreads();
  }
  for (int i = tid; i < NPIX; i += nt) outp[i] = S0[i] / S1[i];
}

// ---------------- final loss ----------------------------------------------
__global__ void zero_accum_kernel(float* accum) {
  if (threadIdx.x < 2) accum[threadIdx.x] = 0.0f;
}
__global__ void loss_kernel(const float* __restrict__ preds, const float* __restrict__ as2,
                            const float* __restrict__ sam, const int* __restrict__ rois,
                            float* __restrict__ accum) {
  int idx = blockIdx.x * blockDim.x + threadIdx.x;
  if (idx >= BATCH * NPIX) return;
  float sg = 1.0f / (1.0f + __expf(-sam[idx]));
  float as = (1.0f - sg + as2[idx]) * 0.5f;
  float roi = (float)rois[idx];
  atomicAdd(&accum[0], roi * fabsf(preds[idx] - as));
  atomicAdd(&accum[1], roi);
}
__global__ void finalize_kernel(const float* __restrict__ accum, float* __restrict__ out) {
  float N = accum[1], s = accum[0];
  out[0] = LOSS_WEIGHT * ((N > 0.0f) ? s / N : s);
}

extern "C" void kernel_launch(void* const* d_in, const int* in_sizes, int n_in,
                              void* d_out, int out_size, void* d_ws, size_t ws_size,
                              hipStream_t stream) {
  (void)in_sizes; (void)n_in; (void)out_size; (void)ws_size;
  const float* preds = (const float*)d_in[0];   // (4,1,80,80)
  const float* low   = (const float*)d_in[1];   // (4,3,320,320)
  const float* high  = (const float*)d_in[2];   // (4,768,80,80)
  const float* sam   = (const float*)d_in[3];   // (4,1,80,80)
  const int*   rois  = (const int*)d_in[4];     // (4,1,80,80)
  float* out = (float*)d_out;

  char* base = (char*)d_ws;
  size_t off = 0;
  auto carve = [&](size_t bytes) -> void* {
    void* r = base + off;
    off += (bytes + 255) & ~(size_t)255;
    return r;
  };
  float* lowr  = (float*)carve((size_t)BATCH * CLOW * NPIX * 4);
  float* w1    = (float*)carve((size_t)BATCH * M_EDGES * 4);
  float* w2    = (float*)carve((size_t)BATCH * M_EDGES * 4);
  int*   mst1  = (int*)carve((size_t)BATCH * M_EDGES * 4);
  int*   mst2  = (int*)carve((size_t)BATCH * M_EDGES * 4);
  int*   par1  = (int*)carve((size_t)BATCH * NPIX * 4);
  int*   dep1  = (int*)carve((size_t)BATCH * NPIX * 4);
  int*   par2  = (int*)carve((size_t)BATCH * NPIX * 4);
  int*   dep2  = (int*)carve((size_t)BATCH * NPIX * 4);
  float* dist1 = (float*)carve((size_t)BATCH * NPIX * 4);
  float* dist2 = (float*)carve((size_t)BATCH * NPIX * 4);
  float* AS1   = (float*)carve((size_t)BATCH * NPIX * 4);
  float* AS2   = (float*)carve((size_t)BATCH * NPIX * 4);
  int*   maxd1 = (int*)carve((size_t)BATCH * 4);
  int*   maxd2 = (int*)carve((size_t)BATCH * 4);
  float* accum = (float*)carve(2 * 4);

  const size_t SH_BOR = 5u * NPIX * 4;          // 128000 B
  const size_t SH_BFS = (2u * NPIX + 8) * 4;    // ~51 KB
  const size_t SH_FLT = 5u * NPIX * 4;          // 128000 B

  resize_low_kernel<<<(BATCH * CLOW * NPIX + 255) / 256, 256, 0, stream>>>(low, lowr);
  w_low_kernel<<<(BATCH * M_EDGES + 255) / 256, 256, 0, stream>>>(lowr, w1);

  // tree 1 (low-level guidance)
  boruvka_kernel<<<BATCH, 1024, SH_BOR, stream>>>(w1, mst1);
  bfs_kernel<<<BATCH, 1024, SH_BFS, stream>>>(mst1, par1, dep1, maxd1);
  dist_low_kernel<<<(BATCH * NPIX + 255) / 256, 256, 0, stream>>>(lowr, par1, dist1);
  treefilter_kernel<<<BATCH, 1024, SH_FLT, stream>>>(preds, dist1, par1, dep1, maxd1, AS1);

  // tree 2 (high-level guidance, 768 channels -> WMMA)
  w_high_wmma_kernel<<<dim3(TILES_E, BATCH), 32, 0, stream>>>(high, w2);
  boruvka_kernel<<<BATCH, 1024, SH_BOR, stream>>>(w2, mst2);
  bfs_kernel<<<BATCH, 1024, SH_BFS, stream>>>(mst2, par2, dep2, maxd2);
  dist_high_wmma_kernel<<<dim3(TILES_N, BATCH), 32, 0, stream>>>(high, par2, dist2);
  treefilter_kernel<<<BATCH, 1024, SH_FLT, stream>>>(AS1, dist2, par2, dep2, maxd2, AS2);

  zero_accum_kernel<<<1, 32, 0, stream>>>(accum);
  loss_kernel<<<(BATCH * NPIX + 255) / 256, 256, 0, stream>>>(preds, AS2, sam, rois, accum);
  finalize_kernel<<<1, 1, 0, stream>>>(accum, out);
}
